// MultiHeadedLatentAttention_15556371546888
// MI455X (gfx1250) — matrix operand order repaired and checked by reading, not compiled, not run on hardware
//
#include <hip/hip_runtime.h>
#include <hip/hip_bf16.h>
#include <math.h>

// ---------------------------------------------------------------------------
// MLA forward for MI455X (gfx1250, wave32).
// All GEMM + attention FLOPs via v_wmma_f32_16x16x32_bf16 (fp32 accum).
// Global->LDS staging uses GLOBAL_LOAD_ASYNC_TO_LDS_B128 (ASYNCcnt) when the
// toolchain exposes the builtin; falls back to register staging otherwise.
// GEMM epilogue is templated on output type so chained GEMMs hand off bf16
// directly (no f32 round-trip through HBM between matmul stages).
// ---------------------------------------------------------------------------

typedef __attribute__((ext_vector_type(16))) __bf16 v16bf;
typedef __attribute__((ext_vector_type(8)))  __bf16 v8bf;
typedef __attribute__((ext_vector_type(8)))  float  v8f;
typedef __attribute__((ext_vector_type(4)))  int    v4i;

#define WMMA_BF16(a, b, c) \
  __builtin_amdgcn_wmma_f32_16x16x32_bf16(false, (a), false, (b), (short)0, (c), false, false)

#if __has_builtin(__builtin_amdgcn_global_load_async_to_lds_b128)
#define USE_ASYNC_LDS 1
#else
#define USE_ASYNC_LDS 0
#endif

// Copy 16 bytes global -> LDS. Async (no VGPR round-trip) when available.
static __device__ __forceinline__ void async_copy16(const __bf16* g, __bf16* l) {
#if USE_ASYNC_LDS
  __builtin_amdgcn_global_load_async_to_lds_b128((v4i*)g, (v4i*)l, 0, 0);
#else
  *(v8bf*)l = *(const v8bf*)g;
#endif
}

static __device__ __forceinline__ void wait_async0() {
#if USE_ASYNC_LDS
#if __has_builtin(__builtin_amdgcn_s_wait_asynccnt)
  __builtin_amdgcn_s_wait_asynccnt(0);
#else
  asm volatile("s_wait_asynccnt 0" ::: "memory");
#endif
#endif
}

// Build a 16-element bf16 fragment from two contiguous 16B chunks (CDNA5
// 16-bit fragment layout: elems 0..7 = K base..base+7, elems 8..15 = +16).
static __device__ __forceinline__ v16bf make_frag(const __bf16* lo, const __bf16* hi) {
  v8bf a0 = *(const v8bf*)lo;
  v8bf a1 = *(const v8bf*)hi;
  v16bf r;
#pragma unroll
  for (int i = 0; i < 8; ++i) { r[i] = a0[i]; r[i + 8] = a1[i]; }
  return r;
}

// ---------------------------------------------------------------------------
// f32 -> bf16 conversion (only used for the input activations)
// ---------------------------------------------------------------------------
__global__ void cvt_f32_bf16_kernel(const float* __restrict__ in,
                                    __bf16* __restrict__ out, size_t n) {
  size_t i = (size_t)blockIdx.x * blockDim.x + threadIdx.x;
  size_t stride = (size_t)gridDim.x * blockDim.x;
  for (; i < n; i += stride) out[i] = (__bf16)in[i];
}

// ---------------------------------------------------------------------------
// Tiled transpose + convert: W (K x N, f32) -> Wt (Npad x K, bf16), zero pad.
// grid (Npad/32, K/32), block (32, 8).
// ---------------------------------------------------------------------------
__global__ __launch_bounds__(256) void transpose_cvt_kernel(
    const float* __restrict__ W, __bf16* __restrict__ Wt, int K, int N) {
  __shared__ float tile[32][33];
  const int n0 = blockIdx.x * 32, k0 = blockIdx.y * 32;
  const int x = threadIdx.x, y = threadIdx.y;
#pragma unroll
  for (int j = 0; j < 4; ++j) {
    int n = n0 + x, k = k0 + y + j * 8;
    tile[y + j * 8][x] = (n < N) ? W[(size_t)k * N + n] : 0.0f;
  }
  __syncthreads();
#pragma unroll
  for (int j = 0; j < 4; ++j) {
    int n = n0 + y + j * 8;
    Wt[(size_t)n * K + k0 + x] = (__bf16)tile[x][y + j * 8];
  }
}

// ---------------------------------------------------------------------------
// bf16 GEMM: C(MxN) = A(MxK,row) * Bt(NxK,row)^T, fp32 accumulate.
// OutT in {float, __bf16}. M%128==0, N%128==0, K%32==0.
// 256 threads = 8 waves in 4(M) x 2(N); each wave 32x64 via 2x4 fragment
// reuse (8 WMMA / K-step). Double-buffered LDS, async global->LDS staging.
// ---------------------------------------------------------------------------
template <typename OutT>
__global__ __launch_bounds__(256) void gemm_bf16_kernel(
    const __bf16* __restrict__ A, const __bf16* __restrict__ Bt,
    OutT* __restrict__ C, int M, int N, int K) {
  __shared__ __bf16 sA[2][128][40];
  __shared__ __bf16 sB[2][128][40];

  const int tid  = threadIdx.x;
  const int lane = tid & 31, wave = tid >> 5;
  const int half = lane >> 4, l16 = lane & 15;
  const int mw = (wave & 3) * 32;
  const int nw = (wave >> 2) * 64;
  const int row0 = blockIdx.y * 128;
  const int col0 = blockIdx.x * 128;

  const int sr = tid >> 1;            // staging row 0..127
  const int sh = (tid & 1) * 16;      // staging K chunk {0,16}
  const __bf16* gA = A  + (size_t)(row0 + sr) * K + sh;
  const __bf16* gB = Bt + (size_t)(col0 + sr) * K + sh;

  v8f acc[8] = {};

  auto stage = [&](int buf, int k0) {
    async_copy16(gA + k0,     &sA[buf][sr][sh]);
    async_copy16(gA + k0 + 8, &sA[buf][sr][sh + 8]);
    async_copy16(gB + k0,     &sB[buf][sr][sh]);
    async_copy16(gB + k0 + 8, &sB[buf][sr][sh + 8]);
  };

  stage(0, 0);
  wait_async0();
  __syncthreads();

  int buf = 0;
  for (int k0 = 0; k0 < K; k0 += 32) {
    if (k0 + 32 < K) stage(buf ^ 1, k0 + 32);

    v16bf af[2], bfr[4];
#pragma unroll
    for (int i = 0; i < 2; ++i)
      af[i] = make_frag(&sA[buf][mw + i * 16 + l16][half * 8],
                        &sA[buf][mw + i * 16 + l16][16 + half * 8]);
#pragma unroll
    for (int j = 0; j < 4; ++j)
      bfr[j] = make_frag(&sB[buf][nw + j * 16 + l16][half * 8],
                         &sB[buf][nw + j * 16 + l16][16 + half * 8]);
#pragma unroll
    for (int i = 0; i < 2; ++i)
#pragma unroll
      for (int j = 0; j < 4; ++j)
        acc[i * 4 + j] = WMMA_BF16(af[i], bfr[j], acc[i * 4 + j]);

    wait_async0();
    __syncthreads();
    buf ^= 1;
  }

  // C/D layout: VGPR r -> row r + 8*half, col = lane&15.
#pragma unroll
  for (int i = 0; i < 2; ++i)
#pragma unroll
    for (int j = 0; j < 4; ++j)
#pragma unroll
      for (int r = 0; r < 8; ++r) {
        int row = row0 + mw + i * 16 + r + half * 8;
        int col = col0 + nw + j * 16 + l16;
        C[(size_t)row * N + col] = (OutT)acc[i * 4 + j][r];
      }
}

// ---------------------------------------------------------------------------
// LayerNorm over first 512 cols of a pitch-`pitch` fp32 row -> bf16
// ---------------------------------------------------------------------------
__global__ __launch_bounds__(256) void layernorm_kernel(
    const float* __restrict__ x, const float* __restrict__ g,
    const float* __restrict__ b, __bf16* __restrict__ out, int pitch) {
  __shared__ float red[256];
  const int row = blockIdx.x, tid = threadIdx.x;
  const float* xr = x + (size_t)row * pitch;
  float x0 = xr[tid], x1 = xr[tid + 256];

  red[tid] = x0 + x1;
  __syncthreads();
  for (int s = 128; s > 0; s >>= 1) { if (tid < s) red[tid] += red[tid + s]; __syncthreads(); }
  float mu = red[0] * (1.0f / 512.0f);
  __syncthreads();

  float d0 = x0 - mu, d1 = x1 - mu;
  red[tid] = d0 * d0 + d1 * d1;
  __syncthreads();
  for (int s = 128; s > 0; s >>= 1) { if (tid < s) red[tid] += red[tid + s]; __syncthreads(); }
  float rs = rsqrtf(red[0] * (1.0f / 512.0f) + 1e-6f);

  out[(size_t)row * 512 + tid]       = (__bf16)(d0 * rs * g[tid] + b[tid]);
  out[(size_t)row * 512 + tid + 256] = (__bf16)(d1 * rs * g[tid + 256] + b[tid + 256]);
}

// ---------------------------------------------------------------------------
// RoPE on q_pe, pack Q (bf16 projection output, pitch 3072) -> (B,H,S,192)
// ---------------------------------------------------------------------------
__global__ __launch_bounds__(192) void rope_q_kernel(
    const __bf16* __restrict__ q, const int* __restrict__ pos_ids,
    const float* __restrict__ cosT, const float* __restrict__ sinT,
    __bf16* __restrict__ Qb, int Sc) {
  const int t = blockIdx.x;
  const int f = threadIdx.x;
  const int b = t / Sc, s = t % Sc;
  const int pos = pos_ids[t];

  float c = 0.f, sn = 0.f, sign = 1.f;
  int pj = 0;
  if (f >= 128) {
    int j = f - 128;
    c = cosT[pos * 64 + j];
    sn = sinT[pos * 64 + j];
    pj = (j < 32) ? j + 32 : j - 32;
    sign = (j < 32) ? -1.f : 1.f;
  }
  for (int h = 0; h < 16; ++h) {
    const __bf16* qh = q + (size_t)t * 3072 + h * 192;
    float v = (float)qh[f];
    if (f >= 128) v = v * c + sign * (float)qh[128 + pj] * sn;
    Qb[(((size_t)b * 16 + h) * Sc + s) * 192 + f] = (__bf16)v;
  }
}

// ---------------------------------------------------------------------------
// Build K = [k_nope | rope(k_pe)] -> (B,H,S,192) and V^T -> (B,H,128,S).
// kv is the bf16 up-projection output (pitch 4096); ckv is f32 pitch 640
// (k_pe at cols 512..575).
// ---------------------------------------------------------------------------
__global__ __launch_bounds__(192) void build_kv_kernel(
    const __bf16* __restrict__ kv, const float* __restrict__ ckv,
    const int* __restrict__ pos_ids,
    const float* __restrict__ cosT, const float* __restrict__ sinT,
    __bf16* __restrict__ Kb, __bf16* __restrict__ Vtb, int Sc) {
  __shared__ __bf16 kpe[64];
  const int t = blockIdx.x, f = threadIdx.x;
  const int b = t / Sc, s = t % Sc;

  if (f >= 128) {
    int j = f - 128;
    int pos = pos_ids[t];
    float c = cosT[pos * 64 + j], sn = sinT[pos * 64 + j];
    const float* kp = ckv + (size_t)t * 640 + 512;
    float partner = (j < 32) ? -kp[j + 32] : kp[j - 32];
    kpe[j] = (__bf16)(kp[j] * c + partner * sn);
  }
  __syncthreads();

  for (int h = 0; h < 16; ++h) {
    size_t krow = (((size_t)b * 16 + h) * Sc + s) * 192;
    if (f < 128) {
      Kb[krow + f] = kv[(size_t)t * 4096 + h * 256 + f];
      Vtb[(((size_t)b * 16 + h) * 128 + f) * Sc + s] =
          kv[(size_t)t * 4096 + h * 256 + 128 + f];
    } else {
      Kb[krow + f] = kpe[f - 128];
    }
  }
}

// ---------------------------------------------------------------------------
// Causal flash attention. Grid: (S/128, H, B). 256 threads = 8 waves; each
// wave owns a 16-row Q tile (Q fragments resident). Key blocks of 32 staged
// in LDS (async). Online softmax in fp32, P recirculated through per-wave LDS.
// ---------------------------------------------------------------------------
__global__ __launch_bounds__(256) void mla_attention_kernel(
    const __bf16* __restrict__ Qb, const __bf16* __restrict__ Kb,
    const __bf16* __restrict__ Vtb, __bf16* __restrict__ Ob, int Sc) {
  __shared__ __bf16 sK[32][200];      // 32 keys x 192 features (+pad)
  __shared__ __bf16 sVt[128][40];     // 128 vd x 32 keys (+pad)
  __shared__ __bf16 sP[8][16][40];    // per-wave P tile (16 x 32)

  const int tid  = threadIdx.x;
  const int lane = tid & 31, wave = tid >> 5;
  const int half = lane >> 4, l16 = lane & 15;
  const int h = blockIdx.y, b = blockIdx.z;
  const int qbase = blockIdx.x * 128 + wave * 16;
  const size_t headQK = (((size_t)b * 16 + h) * Sc) * 192;
  const size_t headV  = (((size_t)b * 16 + h) * 128) * Sc;

  // Resident Q fragments: 6 chunks of K=32 features.
  v16bf aq[6];
  {
    const __bf16* qrow = Qb + headQK + (size_t)(qbase + l16) * 192;
#pragma unroll
    for (int c = 0; c < 6; ++c)
      aq[c] = make_frag(qrow + c * 32 + half * 8, qrow + c * 32 + 16 + half * 8);
  }

  float m_run[8], l_run[8];
  v8f o_acc[8] = {};
#pragma unroll
  for (int r = 0; r < 8; ++r) { m_run[r] = -1e30f; l_run[r] = 0.f; }

  const float scale = rsqrtf(192.0f);
  const int kend = blockIdx.x * 128 + 128;

  // Staging assignments (256 threads)
  const int krow = tid >> 3;          // 0..31
  const int kseg = (tid & 7) * 24;    // 0..168
  const int vrow = tid >> 1;          // 0..127
  const int vo   = (tid & 1) * 16;    // {0,16}

  for (int k0 = 0; k0 < kend; k0 += 32) {
    {
      const __bf16* src = Kb + headQK + (size_t)(k0 + krow) * 192 + kseg;
#pragma unroll
      for (int j = 0; j < 3; ++j)
        async_copy16(src + j * 8, &sK[krow][kseg + j * 8]);
      const __bf16* vs = Vtb + headV + (size_t)vrow * Sc + k0 + vo;
#pragma unroll
      for (int j = 0; j < 2; ++j)
        async_copy16(vs + j * 8, &sVt[vrow][vo + j * 8]);
    }
    wait_async0();
    __syncthreads();

    // Scores: S = Q (16x192) . K^T (192x32) -> two 16x16 accumulators.
    v8f sc[2] = {};
#pragma unroll
    for (int c = 0; c < 6; ++c) {
#pragma unroll
      for (int nt = 0; nt < 2; ++nt) {
        v16bf bk = make_frag(&sK[nt * 16 + l16][c * 32 + half * 8],
                             &sK[nt * 16 + l16][c * 32 + 16 + half * 8]);
        sc[nt] = WMMA_BF16(aq[c], bk, sc[nt]);
      }
    }

    // Scale + causal mask + online softmax (16-lane row reductions).
    float sv[2][8];
#pragma unroll
    for (int nt = 0; nt < 2; ++nt) {
#pragma unroll
      for (int r = 0; r < 8; ++r) {
        int qg = qbase + r + half * 8;
        int kg = k0 + nt * 16 + l16;
        float s = sc[nt][r] * scale;
        sv[nt][r] = (kg > qg) ? -1e30f : s;
      }
    }
#pragma unroll
    for (int r = 0; r < 8; ++r) {
      float mx = fmaxf(sv[0][r], sv[1][r]);
#pragma unroll
      for (int m = 1; m < 16; m <<= 1) mx = fmaxf(mx, __shfl_xor(mx, m, 32));
      float nm    = fmaxf(m_run[r], mx);
      float alpha = __expf(m_run[r] - nm);
      float p0    = __expf(sv[0][r] - nm);
      float p1    = __expf(sv[1][r] - nm);
      float ps    = p0 + p1;
#pragma unroll
      for (int m = 1; m < 16; m <<= 1) ps += __shfl_xor(ps, m, 32);
      l_run[r] = l_run[r] * alpha + ps;
      m_run[r] = nm;
#pragma unroll
      for (int vt = 0; vt < 8; ++vt) o_acc[vt][r] *= alpha;
      sP[wave][r + half * 8][l16]      = (__bf16)p0;
      sP[wave][r + half * 8][16 + l16] = (__bf16)p1;
    }
    // Wave-local LDS store->load ordering before reading P in A-layout.
    asm volatile("s_wait_dscnt 0" ::: "memory");

    v16bf pa = make_frag(&sP[wave][l16][half * 8], &sP[wave][l16][16 + half * 8]);
#pragma unroll
    for (int vt = 0; vt < 8; ++vt) {
      v16bf bv = make_frag(&sVt[vt * 16 + l16][half * 8],
                           &sVt[vt * 16 + l16][16 + half * 8]);
      o_acc[vt] = WMMA_BF16(pa, bv, o_acc[vt]);
    }
    __syncthreads();
  }

  // Normalize and pack to (b, s, h*128+d) bf16 for the W_o GEMM.
#pragma unroll
  for (int r = 0; r < 8; ++r) {
    float inv = 1.0f / l_run[r];
    int qg = qbase + r + half * 8;
    size_t rowbase = ((size_t)b * Sc + qg) * 2048 + h * 128;
#pragma unroll
    for (int vt = 0; vt < 8; ++vt)
      Ob[rowbase + vt * 16 + l16] = (__bf16)(o_acc[vt][r] * inv);
  }
}

// ---------------------------------------------------------------------------
// Host orchestration
// ---------------------------------------------------------------------------
extern "C" void kernel_launch(void* const* d_in, const int* in_sizes, int n_in,
                              void* d_out, int out_size, void* d_ws, size_t ws_size,
                              hipStream_t stream) {
  (void)in_sizes; (void)n_in; (void)out_size; (void)ws_size;

  const float* hidden = (const float*)d_in[0];
  const int*   pos    = (const int*)d_in[1];
  const float* cosT   = (const float*)d_in[2];
  const float* sinT   = (const float*)d_in[3];
  const float* W_q    = (const float*)d_in[4];
  const float* W_kva  = (const float*)d_in[5];
  const float* ln_g   = (const float*)d_in[6];
  const float* ln_b   = (const float*)d_in[7];
  const float* W_kvb  = (const float*)d_in[8];
  const float* W_o    = (const float*)d_in[9];
  float* out = (float*)d_out;

  const int Bc = 2, Sc = 2048, HID = 2048, Hh = 16;
  const int T = Bc * Sc;  // 4096 tokens

  char* ws = (char*)d_ws;
  size_t off = 0;
  auto alloc = [&](size_t bytes) -> void* {
    void* p = (void*)(ws + off);
    off += (bytes + 255) & ~(size_t)255;
    return p;
  };

  __bf16* Xb      = (__bf16*)alloc((size_t)T * HID * 2);
  __bf16* Wq_t    = (__bf16*)alloc((size_t)3072 * HID * 2);   // (N x K)
  __bf16* Wkva_t  = (__bf16*)alloc((size_t)640 * HID * 2);    // padded N=640
  __bf16* Wkvb_t  = (__bf16*)alloc((size_t)4096 * 512 * 2);
  __bf16* Wo_t    = (__bf16*)alloc((size_t)HID * HID * 2);
  __bf16* qfb     = (__bf16*)alloc((size_t)T * 3072 * 2);     // Q proj (bf16)
  float*  ckv     = (float*)alloc((size_t)T * 640 * 4);       // f32 for LN stats
  __bf16* lnb16   = (__bf16*)alloc((size_t)T * 512 * 2);
  __bf16* kvb     = (__bf16*)alloc((size_t)T * 4096 * 2);     // KV proj (bf16)
  __bf16* Qb      = (__bf16*)alloc((size_t)T * Hh * 192 * 2);
  __bf16* Kb      = (__bf16*)alloc((size_t)T * Hh * 192 * 2);
  __bf16* Vtb     = (__bf16*)alloc((size_t)T * Hh * 128 * 2);
  __bf16* Ab      = (__bf16*)alloc((size_t)T * HID * 2);
  __bf16* tmpb    = (__bf16*)alloc((size_t)T * HID * 2);

  auto cvt = [&](const float* src, __bf16* dst, size_t n) {
    int blocks = (int)((n + 1023) / 1024);
    cvt_f32_bf16_kernel<<<blocks, 256, 0, stream>>>(src, dst, n);
  };
  auto tcvt = [&](const float* W, __bf16* Wt, int K, int N, int Npad) {
    transpose_cvt_kernel<<<dim3(Npad / 32, K / 32), dim3(32, 8), 0, stream>>>(W, Wt, K, N);
  };
  auto gemm_f32 = [&](const __bf16* A, const __bf16* Bt, float* C, int M, int N, int K) {
    gemm_bf16_kernel<float><<<dim3(N / 128, M / 128), 256, 0, stream>>>(A, Bt, C, M, N, K);
  };
  auto gemm_b16 = [&](const __bf16* A, const __bf16* Bt, __bf16* C, int M, int N, int K) {
    gemm_bf16_kernel<__bf16><<<dim3(N / 128, M / 128), 256, 0, stream>>>(A, Bt, C, M, N, K);
  };

  // 1) Precision staging (+ one-time weight transposes so GEMM stages with
  //    pure b128 traffic, no in-kernel scalar transpose)
  cvt(hidden, Xb, (size_t)T * HID);
  tcvt(W_q,   Wq_t,   HID, 3072, 3072);
  tcvt(W_kva, Wkva_t, HID, 576,  640);
  tcvt(W_kvb, Wkvb_t, 512, 4096, 4096);
  tcvt(W_o,   Wo_t,   HID, 2048, 2048);

  // 2) Projections (bf16 outputs where the consumer is another WMMA stage)
  gemm_b16(Xb, Wq_t,   qfb, T, 3072, HID);
  gemm_f32(Xb, Wkva_t, ckv, T, 640,  HID);   // f32: LayerNorm stats next
  layernorm_kernel<<<T, 256, 0, stream>>>(ckv, ln_g, ln_b, lnb16, 640);
  gemm_b16(lnb16, Wkvb_t, kvb, T, 4096, 512);

  // 3) RoPE + attention operand packing
  rope_q_kernel<<<T, 192, 0, stream>>>(qfb, pos, cosT, sinT, Qb, Sc);
  build_kv_kernel<<<T, 192, 0, stream>>>(kvb, ckv, pos, cosT, sinT, Kb, Vtb, Sc);

  // 4) Causal flash attention (WMMA QK^T + WMMA PV, fp32 online softmax)
  mla_attention_kernel<<<dim3(Sc / 128, Hh, Bc), 256, 0, stream>>>(Qb, Kb, Vtb, Ab, Sc);

  // 5) Output projection, applied twice (per reference); bf16 hand-off
  gemm_b16(Ab, Wo_t, tmpb, T, HID, HID);
  gemm_f32(tmpb, Wo_t, out, T, HID, HID);
}
